// Block_34840774705280
// MI455X (gfx1250) — compile-verified, matrix operand order checked by reading, not run on hardware
//
#include <hip/hip_runtime.h>
#include <hip/hip_bf16.h>
#include <math.h>

// ---------------------------------------------------------------------------
// CDNA5 (gfx1250) deformable-attention transformer block.
// GEMMs: v_wmma_f32_16x16x32_bf16, wave32, B staged in LDS (WGP 320KB),
// fully unrolled K loop (KS is a template parameter), fp32 accumulate.
// ---------------------------------------------------------------------------

typedef __attribute__((ext_vector_type(16))) __bf16 v16bf;
typedef __attribute__((ext_vector_type(8)))  float  v8f;

#define DIMC 384
#define HIDC 1536
#define NHC  6
#define NPC  4

__device__ __forceinline__ unsigned short f2bf(float f) {
  union { float f; unsigned u; } c; c.f = f;
  unsigned u = c.u;
  u += 0x7fffu + ((u >> 16) & 1u);       // round-to-nearest-even
  return (unsigned short)(u >> 16);
}

// Packed WMMA fragment layout (matches ISA 16-bit 16x32 A-matrix layout):
// ushort index = ((tile*KS + ks)*32 + lane)*16 + e
// lane = (row%16) + 16*half, half = bit3 of (k%32).
__device__ __forceinline__ void packStore(unsigned short* __restrict__ dst,
                                          int KS, int m, int k, unsigned short v) {
  int kk = k & 31, ks = k >> 5;
  int half = (kk >> 3) & 1;
  int kl = kk - 8 * half;                 // {0..7, 16..23}
  int e = (kl < 8) ? kl : (kl - 8);
  int lane = (m & 15) + 16 * half;
  dst[((size_t)((m >> 4) * KS + ks) * 32 + lane) * 16 + e] = v;
}

// ------------------------- LayerNorm (wave per row) -------------------------
__global__ __launch_bounds__(256) void ln_pack_kernel(
    const float* __restrict__ x, const float* __restrict__ w,
    const float* __restrict__ b, float* __restrict__ outF,
    unsigned short* __restrict__ outPack, int M) {
  int wave = threadIdx.x >> 5;
  int lane = threadIdx.x & 31;
  int row = blockIdx.x * 8 + wave;
  if (row >= M) return;
  const float* xr = x + (size_t)row * DIMC;
  float vals[12];
  float s = 0.f;
  #pragma unroll
  for (int i = 0; i < 12; ++i) { vals[i] = xr[lane + 32 * i]; s += vals[i]; }
  #pragma unroll
  for (int off = 16; off; off >>= 1) s += __shfl_xor(s, off, 32);
  float mu = s * (1.0f / DIMC);
  float vs = 0.f;
  #pragma unroll
  for (int i = 0; i < 12; ++i) { float d = vals[i] - mu; vs += d * d; }
  #pragma unroll
  for (int off = 16; off; off >>= 1) vs += __shfl_xor(vs, off, 32);
  float rstd = rsqrtf(vs * (1.0f / DIMC) + 1e-5f);
  #pragma unroll
  for (int i = 0; i < 12; ++i) {
    int c = lane + 32 * i;
    float h = (vals[i] - mu) * rstd * w[c] + b[c];
    if (outF) outF[(size_t)row * DIMC + c] = h;
    packStore(outPack, DIMC / 32, row, c, f2bf(h));
  }
}

// ------------------- weight fp32 (KxN row-major) -> packed bf16 ------------
__global__ __launch_bounds__(256) void pack_w_kernel(
    const float* __restrict__ W, unsigned short* __restrict__ P, int K, int N) {
  int idx = blockIdx.x * 256 + threadIdx.x;
  if (idx >= K * N) return;
  int e    = idx & 15;
  int lane = (idx >> 4) & 31;
  int rest = idx >> 9;
  int KS = K >> 5;
  int ks = rest % KS;
  int tile = rest / KS;
  int half = lane >> 4;
  int kl = (e < 8) ? e : e + 8;
  int k = ks * 32 + kl + 8 * half;
  int n = tile * 16 + (lane & 15);
  P[idx] = f2bf(W[(size_t)k * N + n]);
}

// ------------------------- WMMA GEMM: D = A*B (+epilogue) ------------------
// Block: 256 threads = 8 wave32. Block tile 256(M) x 64(N).
// B for the whole K range staged once into LDS, shared by all 8 waves.
// Each wave owns two 16-row M-tiles (8 v8f accumulators).
// K loop fully unrolled (KS compile-time) so the scheduler hoists A/B loads
// across WMMAs and renames registers (no ping-pong moves, batched waits).
// MODE 0: store f32            (v projection)
// MODE 1: +bias +resid -> f32  (proj + residual)
// MODE 2: +bias, GELU -> packed bf16 (fc1)
// MODE 3: +bias +resid -> f32  (fc2 + residual -> output)
template <int MODE, int KS>
__global__ __launch_bounds__(256) void gemm_wmma_kernel(
    const unsigned short* __restrict__ Apack, const unsigned short* __restrict__ Bpack,
    int M, int N,
    const float* __restrict__ bias, const float* __restrict__ resid,
    float* __restrict__ outF, unsigned short* __restrict__ outPack, int outPackKS) {
  extern __shared__ unsigned short ldsB[];   // KS * 4 * 32 * 16 ushorts
  int lane = threadIdx.x & 31;
  int wave = threadIdx.x >> 5;
  int tileN0 = blockIdx.x * 4;               // 4 x 16 = 64 columns per block
  int tileM0 = blockIdx.y * 16;              // 16 m-tiles = 256 rows per block

  // ---- cooperative B stage: global (packed bf16) -> LDS, b128 granules ----
  {
    const uint4* Bg = (const uint4*)Bpack;   // fragment = 2 x uint4 (32B)
    uint4* Ls = (uint4*)ldsB;
    #pragma unroll 4
    for (int u = threadIdx.x; u < KS * 128; u += 256) {
      int l = u & 31, j = (u >> 5) & 3, ks = u >> 7;
      size_t gf = ((size_t)(tileN0 + j) * KS + ks) * 32 + l;
      Ls[2 * u]     = Bg[2 * gf];
      Ls[2 * u + 1] = Bg[2 * gf + 1];
    }
  }
  __syncthreads();

  int tileMa = tileM0 + wave * 2;
  int tileMb = tileMa + 1;
  const v16bf* __restrict__ Ap = reinterpret_cast<const v16bf*>(Apack);
  const v16bf* __restrict__ Bl = reinterpret_cast<const v16bf*>(ldsB);

  v8f acc[8] = {};
  #pragma unroll
  for (int ks = 0; ks < KS; ++ks) {
    // A fragments for this k-step (global, L2-resident, hoisted by scheduler)
    v16bf a0 = Ap[((size_t)tileMa * KS + ks) * 32 + lane];
    v16bf a1 = Ap[((size_t)tileMb * KS + ks) * 32 + lane];
    // All 4 B fragments first: one ds wait covers 8 WMMAs
    v16bf b0 = Bl[(ks * 4 + 0) * 32 + lane];
    v16bf b1 = Bl[(ks * 4 + 1) * 32 + lane];
    v16bf b2 = Bl[(ks * 4 + 2) * 32 + lane];
    v16bf b3 = Bl[(ks * 4 + 3) * 32 + lane];
    acc[0] = __builtin_amdgcn_wmma_f32_16x16x32_bf16(false, a0, false, b0, (short)0, acc[0], false, false);
    acc[4] = __builtin_amdgcn_wmma_f32_16x16x32_bf16(false, a1, false, b0, (short)0, acc[4], false, false);
    acc[1] = __builtin_amdgcn_wmma_f32_16x16x32_bf16(false, a0, false, b1, (short)0, acc[1], false, false);
    acc[5] = __builtin_amdgcn_wmma_f32_16x16x32_bf16(false, a1, false, b1, (short)0, acc[5], false, false);
    acc[2] = __builtin_amdgcn_wmma_f32_16x16x32_bf16(false, a0, false, b2, (short)0, acc[2], false, false);
    acc[6] = __builtin_amdgcn_wmma_f32_16x16x32_bf16(false, a1, false, b2, (short)0, acc[6], false, false);
    acc[3] = __builtin_amdgcn_wmma_f32_16x16x32_bf16(false, a0, false, b3, (short)0, acc[3], false, false);
    acc[7] = __builtin_amdgcn_wmma_f32_16x16x32_bf16(false, a1, false, b3, (short)0, acc[7], false, false);
  }

  // C/D layout: VGPR r holds (m0 + r + 8*(lane>>4), tileN*16 + (lane&15))
  const int N_ = N;
  #pragma unroll
  for (int half = 0; half < 2; ++half) {
    int m0 = (half ? tileMb : tileMa) * 16;
    #pragma unroll
    for (int j = 0; j < 4; ++j) {
      int n = (tileN0 + j) * 16 + (lane & 15);
      #pragma unroll
      for (int r = 0; r < 8; ++r) {
        int m = m0 + r + 8 * (lane >> 4);
        float v = acc[half * 4 + j][r];
        if constexpr (MODE == 1 || MODE == 3) {
          v += bias[n] + resid[(size_t)m * N_ + n];
          outF[(size_t)m * N_ + n] = v;
        } else if constexpr (MODE == 2) {
          v += bias[n];
          v = 0.5f * v * (1.0f + erff(v * 0.70710678118f));   // exact GELU
          packStore(outPack, outPackKS, m, n, f2bf(v));
        } else {
          outF[(size_t)m * N_ + n] = v;
        }
      }
    }
  }
}

// ----------- deformable sampling: one thread per (token, head) -------------
__global__ __launch_bounds__(256) void attn_sample_kernel(
    const float* __restrict__ hF, const float* __restrict__ vF,
    const float* __restrict__ aw_w, const float* __restrict__ aw_b,
    const float* __restrict__ off_w, const float* __restrict__ off_b,
    unsigned short* __restrict__ aPack, int M) {
  int t = blockIdx.x * 256 + threadIdx.x;
  if (t >= M * NHC) return;
  int head = t / M;
  int token = t - head * M;
  int b = token >> 12;                     // N = 64*64 = 4096 tokens/image
  int pos = token & 4095;
  int py = pos >> 6, px = pos & 63;

  const float* hr = hF + (size_t)token * DIMC;
  int c4 = head * NPC;
  float lg[4] = {0.f, 0.f, 0.f, 0.f};
  float of[8] = {0.f, 0.f, 0.f, 0.f, 0.f, 0.f, 0.f, 0.f};
  for (int k = 0; k < DIMC; ++k) {
    float hv = hr[k];
    const float* awr = aw_w + (size_t)k * (NHC * NPC) + c4;
    const float* ofr = off_w + (size_t)k * (NHC * NPC * 2) + c4 * 2;
    #pragma unroll
    for (int p = 0; p < 4; ++p) lg[p] = fmaf(hv, awr[p], lg[p]);
    #pragma unroll
    for (int i = 0; i < 8; ++i) of[i] = fmaf(hv, ofr[i], of[i]);
  }
  #pragma unroll
  for (int p = 0; p < 4; ++p) lg[p] += aw_b[c4 + p];
  #pragma unroll
  for (int i = 0; i < 8; ++i) of[i] += off_b[c4 * 2 + i];

  float mx = fmaxf(fmaxf(lg[0], lg[1]), fmaxf(lg[2], lg[3]));
  float wgt[4], se = 0.f;
  #pragma unroll
  for (int p = 0; p < 4; ++p) { wgt[p] = __expf(lg[p] - mx); se += wgt[p]; }
  float inv = 1.0f / se;
  #pragma unroll
  for (int p = 0; p < 4; ++p) wgt[p] *= inv;

  float out[64];
  #pragma unroll
  for (int d = 0; d < 64; ++d) out[d] = 0.f;

  const float* vbase = vF + (size_t)b * 4096 * DIMC + head * 64;
  for (int p = 0; p < 4; ++p) {
    float gx = (float)px + of[2 * p];
    float gy = (float)py + of[2 * p + 1];
    int x0 = (int)floorf(gx), y0 = (int)floorf(gy);
    float wx1 = gx - (float)x0, wy1 = gy - (float)y0;
    float wx0 = 1.f - wx1, wy0 = 1.f - wy1;
    int xs[2] = {x0, x0 + 1}, ys[2] = {y0, y0 + 1};
    float wxs[2] = {wx0, wx1}, wys[2] = {wy0, wy1};
    #pragma unroll
    for (int cy = 0; cy < 2; ++cy) {
      #pragma unroll
      for (int cx = 0; cx < 2; ++cx) {
        int xi = xs[cx], yi = ys[cy];
        float w = wxs[cx] * wys[cy] * wgt[p];
        if (xi >= 0 && xi < 64 && yi >= 0 && yi < 64) {
          const float* vr = vbase + (size_t)(yi * 64 + xi) * DIMC;
          #pragma unroll
          for (int d = 0; d < 64; d += 4) {
            float4 q = *(const float4*)(vr + d);
            out[d]     = fmaf(w, q.x, out[d]);
            out[d + 1] = fmaf(w, q.y, out[d + 1]);
            out[d + 2] = fmaf(w, q.z, out[d + 2]);
            out[d + 3] = fmaf(w, q.w, out[d + 3]);
          }
        }
      }
    }
  }
  int cbase = head * 64;
  #pragma unroll
  for (int d = 0; d < 64; ++d)
    packStore(aPack, DIMC / 32, token, cbase + d, f2bf(out[d]));
}

// ---------------------------------------------------------------------------
extern "C" void kernel_launch(void* const* d_in, const int* in_sizes, int n_in,
                              void* d_out, int out_size, void* d_ws, size_t ws_size,
                              hipStream_t stream) {
  const float* x      = (const float*)d_in[0];
  const float* n1_w   = (const float*)d_in[3];
  const float* n1_b   = (const float*)d_in[4];
  const float* v_w    = (const float*)d_in[5];
  const float* aw_w   = (const float*)d_in[6];
  const float* aw_b   = (const float*)d_in[7];
  const float* off_w  = (const float*)d_in[8];
  const float* off_b  = (const float*)d_in[9];
  const float* proj_w = (const float*)d_in[10];
  const float* proj_b = (const float*)d_in[11];
  const float* n2_w   = (const float*)d_in[12];
  const float* n2_b   = (const float*)d_in[13];
  const float* fc1_w  = (const float*)d_in[14];
  const float* fc1_b  = (const float*)d_in[15];
  const float* fc2_w  = (const float*)d_in[16];
  const float* fc2_b  = (const float*)d_in[17];

  const int M = in_sizes[0] / DIMC;        // B * H * W = 32768

  // ---- deterministic workspace carve (256B aligned) ----
  char* ws = (char*)d_ws;
  size_t o = 0;
  auto carve = [&](size_t bytes) {
    void* p = ws + o;
    o = (o + bytes + 255) & ~(size_t)255;
    return p;
  };
  float*          hF     = (float*)carve((size_t)M * DIMC * 4);
  unsigned short* hPack  = (unsigned short*)carve((size_t)M * DIMC * 2);
  float*          vF     = (float*)carve((size_t)M * DIMC * 4);
  unsigned short* aPack  = (unsigned short*)carve((size_t)M * DIMC * 2);
  float*          x1F    = (float*)carve((size_t)M * DIMC * 4);
  unsigned short* h2Pack = (unsigned short*)carve((size_t)M * DIMC * 2);
  unsigned short* gPack  = (unsigned short*)carve((size_t)M * HIDC * 2);
  unsigned short* vwP    = (unsigned short*)carve((size_t)DIMC * DIMC * 2);
  unsigned short* projP  = (unsigned short*)carve((size_t)DIMC * DIMC * 2);
  unsigned short* fc1P   = (unsigned short*)carve((size_t)DIMC * HIDC * 2);
  unsigned short* fc2P   = (unsigned short*)carve((size_t)HIDC * DIMC * 2);
  (void)ws_size; (void)n_in; (void)out_size;

  const size_t ldsK384  = (size_t)(DIMC / 32) * 4096;   // 48 KB
  const size_t ldsK1536 = (size_t)(HIDC / 32) * 4096;   // 192 KB (WGP has 320 KB)

  // ---- weight packing ----
  pack_w_kernel<<<(DIMC * DIMC + 255) / 256, 256, 0, stream>>>(v_w, vwP, DIMC, DIMC);
  pack_w_kernel<<<(DIMC * DIMC + 255) / 256, 256, 0, stream>>>(proj_w, projP, DIMC, DIMC);
  pack_w_kernel<<<(DIMC * HIDC + 255) / 256, 256, 0, stream>>>(fc1_w, fc1P, DIMC, HIDC);
  pack_w_kernel<<<(HIDC * DIMC + 255) / 256, 256, 0, stream>>>(fc2_w, fc2P, HIDC, DIMC);

  // ---- LN1 -> h (f32 for sampling dots) + packed bf16 A fragments ----
  ln_pack_kernel<<<(M + 7) / 8, 256, 0, stream>>>(x, n1_w, n1_b, hF, hPack, M);

  // ---- v = h @ v_w  (row-major f32 out doubles as (B*NH,HW,Dh)) ----
  gemm_wmma_kernel<0, DIMC / 32><<<dim3(DIMC / 64, M / 256), 256, ldsK384, stream>>>(
      hPack, vwP, M, DIMC, nullptr, nullptr, vF, nullptr, 0);

  // ---- offsets/weights + softmax + bilinear sample -> aPack (bf16) ----
  attn_sample_kernel<<<(M * NHC + 255) / 256, 256, 0, stream>>>(
      hF, vF, aw_w, aw_b, off_w, off_b, aPack, M);

  // ---- x1 = x + a @ proj_w + proj_b ----
  gemm_wmma_kernel<1, DIMC / 32><<<dim3(DIMC / 64, M / 256), 256, ldsK384, stream>>>(
      aPack, projP, M, DIMC, proj_b, x, x1F, nullptr, 0);

  // ---- LN2 -> packed bf16 only ----
  ln_pack_kernel<<<(M + 7) / 8, 256, 0, stream>>>(x1F, n2_w, n2_b, nullptr, h2Pack, M);

  // ---- fc1 + GELU -> gPack (bf16, packed for fc2) ----
  gemm_wmma_kernel<2, DIMC / 32><<<dim3(HIDC / 64, M / 256), 256, ldsK384, stream>>>(
      h2Pack, fc1P, M, HIDC, fc1_b, nullptr, nullptr, gPack, HIDC / 32);

  // ---- out = x1 + g @ fc2_w + fc2_b ----
  gemm_wmma_kernel<3, HIDC / 32><<<dim3(DIMC / 64, M / 256), 256, ldsK1536, stream>>>(
      gPack, fc2P, M, DIMC, fc2_b, x1F, (float*)d_out, nullptr, 0);
}
